// GIN_352187318576
// MI455X (gfx1250) — matrix operand order repaired and checked by reading, not compiled, not run on hardware
//
#include <hip/hip_runtime.h>

typedef __attribute__((ext_vector_type(16))) _Float16 v16h;
typedef __attribute__((ext_vector_type(8)))  _Float16 v8h;
typedef __attribute__((ext_vector_type(8)))  float    v8f;

static constexpr int kNodes = 50000;
static constexpr int kEdges = 800000;
static constexpr int kHid   = 128;

// ---------------- kernel: zero h_neigh accumulator ----------------
__global__ void zero_f4(float4* __restrict__ p, int n4) {
  int i = blockIdx.x * blockDim.x + threadIdx.x;
  if (i < n4) p[i] = make_float4(0.f, 0.f, 0.f, 0.f);
}

// ---------------- kernel: convert the 4 weight matrices to f16 ----------------
// Packed layout (halves): W1_0 [0,16384) W2_0 [16384,32768) W1_1 [32768,57344) W2_1 [57344,73728)
__global__ void cvt_weights(const float* __restrict__ W10, const float* __restrict__ W20,
                            const float* __restrict__ W11, const float* __restrict__ W21,
                            _Float16* __restrict__ out) {
  int i = blockIdx.x * blockDim.x + threadIdx.x;
  if (i < 16384)       out[i] = (_Float16)W10[i];
  else if (i < 32768)  out[i] = (_Float16)W20[i - 16384];
  else if (i < 57344)  out[i] = (_Float16)W11[i - 32768];
  else if (i < 73728)  out[i] = (_Float16)W21[i - 57344];
}

// ---------------- kernel: scatter-add edge features ----------------
// One thread per (edge, 4-float group): 800000 * 16 threads, float4 loads,
// relaxed agent-scope f32 atomics -> global_atomic_add_f32 into L2-resident array.
__global__ void scatter_edges(const float4* __restrict__ ef4,
                              const int* __restrict__ dst,
                              float* __restrict__ hneigh) {
  long long t = (long long)blockIdx.x * blockDim.x + threadIdx.x;
  int e = (int)(t >> 4);
  int g = (int)(t & 15);
  if (e >= kEdges) return;
  float4 v = ef4[(long long)e * 16 + g];
  int d = dst[e];
  float* b = hneigh + (long long)d * 64 + g * 4;
  __hip_atomic_fetch_add(b + 0, v.x, __ATOMIC_RELAXED, __HIP_MEMORY_SCOPE_AGENT);
  __hip_atomic_fetch_add(b + 1, v.y, __ATOMIC_RELAXED, __HIP_MEMORY_SCOPE_AGENT);
  __hip_atomic_fetch_add(b + 2, v.z, __ATOMIC_RELAXED, __HIP_MEMORY_SCOPE_AGENT);
  __hip_atomic_fetch_add(b + 3, v.w, __ATOMIC_RELAXED, __HIP_MEMORY_SCOPE_AGENT);
}

// ---------------- fused MLP: WMMA helpers ----------------
union AF { v16h v; v8h h8[2]; };

// One wave computes a 16x16 f32 tile of  X[32 x K] @ W[K x 128]  (X in LDS f16,
// W in global f16 row-major). A-fragment per CDNA5 16-bit 16x32 layout:
//   lanes 0-15 : row M=lane,    K runs {kb..kb+7, kb+16..kb+23}
//   lanes 16-31: row M=lane-16, K runs {kb+8..kb+15, kb+24..kb+31}
// B-fragment: column N = n_tile*16 + (lane&15), K runs same as A-fragment.
template <int K, int LDX>
__device__ inline v8f wmma_row_tile(const _Float16* __restrict__ xin,
                                    const _Float16* __restrict__ W,
                                    int lane, int m_sub, int n_tile) {
  v8f c = {};
  const _Float16* arow = xin + (m_sub * 16 + (lane & 15)) * LDX;
  const int koff = (lane >> 4) << 3;  // 0 or 8
#pragma unroll
  for (int kb = 0; kb < K; kb += 32) {
    AF a, b;
    a.h8[0] = *(const v8h*)(arow + kb + koff);
    a.h8[1] = *(const v8h*)(arow + kb + 16 + koff);
    const _Float16* bp = W + (kb + koff) * kHid + n_tile * 16 + (lane & 15);
#pragma unroll
    for (int i = 0; i < 8; ++i) {
      b.v[i]     = bp[i * kHid];
      b.v[8 + i] = bp[(16 + i) * kHid];
    }
    c = __builtin_amdgcn_wmma_f32_16x16x32_f16(false, a.v, false, b.v,
                                               (short)0, c, false, false);
  }
  return c;
}

// C/D layout: VGPR r -> (lanes 0-15: M=r, N=lane) (lanes 16-31: M=8+r, N=lane-16)
__device__ inline void relu_store_lds(_Float16* __restrict__ xout, int ldo, v8f c,
                                      float bv, int lane, int m_sub, int n) {
  const int mb = m_sub * 16 + ((lane < 16) ? 0 : 8);
#pragma unroll
  for (int r = 0; r < 8; ++r) {
    float v = c[r] + bv;
    xout[(mb + r) * ldo + n] = (_Float16)(v > 0.f ? v : 0.f);
  }
}

// ---------------- fused 4-GEMM MLP over 32 nodes per block ----------------
__global__ __launch_bounds__(512)
void gin_fused(const float* __restrict__ nfeats,
               const float* __restrict__ hneigh,
               const _Float16* __restrict__ Wh,
               const float* __restrict__ b10, const float* __restrict__ b20,
               const float* __restrict__ b11, const float* __restrict__ b21,
               float* __restrict__ out) {
  __shared__ _Float16 bufA[32 * 128];  // [nfeats | hneigh] f16
  __shared__ _Float16 bufB[32 * 128];  // hidden activations
  __shared__ _Float16 bufC[32 * 192];  // [h | hneigh] for layer 1

  const int tid    = threadIdx.x;
  const int lane   = tid & 31;
  const int wave   = tid >> 5;
  const int m_sub  = wave >> 3;  // 0..1  -> node rows 0-15 / 16-31
  const int n_tile = wave & 7;   // 0..7  -> output cols n_tile*16..+15
  const int node0  = blockIdx.x * 32;

  const _Float16* W10h = Wh;
  const _Float16* W20h = Wh + 16384;
  const _Float16* W11h = Wh + 32768;
  const _Float16* W21h = Wh + 57344;

  // Stage node inputs into LDS (zero-pad rows past kNodes).
  for (int idx = tid; idx < 32 * 64; idx += 512) {
    int r = idx >> 6, col = idx & 63;
    int node = node0 + r;
    float nf = 0.f, hn = 0.f;
    if (node < kNodes) {
      nf = nfeats[(long long)node * 64 + col];
      hn = hneigh[(long long)node * 64 + col];
    }
    bufA[r * 128 + col]        = (_Float16)nf;
    bufA[r * 128 + 64 + col]   = (_Float16)hn;
    bufC[r * 192 + 128 + col]  = (_Float16)hn;  // concat slot for layer 1
  }
  __syncthreads();

  const int n = n_tile * 16 + (lane & 15);

  // Layer 0, GEMM 1: [32x128] @ [128x128]
  {
    v8f c = wmma_row_tile<128, 128>(bufA, W10h, lane, m_sub, n_tile);
    relu_store_lds(bufB, 128, c, b10[n], lane, m_sub, n);
  }
  __syncthreads();

  // Layer 0, GEMM 2: [32x128] @ [128x128] -> bufC[:, 0:128]
  {
    v8f c = wmma_row_tile<128, 128>(bufB, W20h, lane, m_sub, n_tile);
    relu_store_lds(bufC, 192, c, b20[n], lane, m_sub, n);
  }
  __syncthreads();

  // Layer 1, GEMM 1: [32x192] @ [192x128]
  {
    v8f c = wmma_row_tile<192, 192>(bufC, W11h, lane, m_sub, n_tile);
    relu_store_lds(bufB, 128, c, b11[n], lane, m_sub, n);
  }
  __syncthreads();

  // Layer 1, GEMM 2: [32x128] @ [128x128] -> global f32
  {
    v8f c = wmma_row_tile<128, 128>(bufB, W21h, lane, m_sub, n_tile);
    float bv = b21[n];
    const int mb = m_sub * 16 + ((lane < 16) ? 0 : 8);
#pragma unroll
    for (int r = 0; r < 8; ++r) {
      int node = node0 + mb + r;
      if (node < kNodes) {
        float v = c[r] + bv;
        out[(long long)node * 128 + n] = (v > 0.f ? v : 0.f);
      }
    }
  }
}

extern "C" void kernel_launch(void* const* d_in, const int* in_sizes, int n_in,
                              void* d_out, int out_size, void* d_ws, size_t ws_size,
                              hipStream_t stream) {
  const float* nfeats = (const float*)d_in[0];
  const float* efeats = (const float*)d_in[1];
  const int*   dst    = (const int*)d_in[2];
  const float* W10 = (const float*)d_in[3];
  const float* b10 = (const float*)d_in[4];
  const float* W20 = (const float*)d_in[5];
  const float* b20 = (const float*)d_in[6];
  const float* W11 = (const float*)d_in[7];
  const float* b11 = (const float*)d_in[8];
  const float* W21 = (const float*)d_in[9];
  const float* b21 = (const float*)d_in[10];
  float* out = (float*)d_out;

  // Workspace layout: h_neigh f32 [50000*64] then packed f16 weights [73728].
  float*    hneigh = (float*)d_ws;
  _Float16* Wh     = (_Float16*)((char*)d_ws + (size_t)kNodes * 64 * sizeof(float));

  // 1) zero the scatter accumulator (d_ws is poisoned, must re-zero every call)
  {
    int n4 = kNodes * 64 / 4;  // 800000
    zero_f4<<<(n4 + 255) / 256, 256, 0, stream>>>((float4*)hneigh, n4);
  }
  // 2) weights -> f16
  cvt_weights<<<(73728 + 255) / 256, 256, 0, stream>>>(W10, W20, W11, W21, Wh);
  // 3) scatter-add edge features (bandwidth-dominant: ~205 MB read)
  {
    long long nthreads = (long long)kEdges * 16;
    scatter_edges<<<(int)(nthreads / 256), 256, 0, stream>>>(
        (const float4*)efeats, dst, hneigh);
  }
  // 4) fused WMMA MLP, 32 nodes / 512 threads per block
  gin_fused<<<(kNodes + 31) / 32, 512, 0, stream>>>(
      nfeats, hneigh, Wh, b10, b20, b11, b21, out);
}
